// SelfAttentionLayer_14894946583181
// MI455X (gfx1250) — matrix operand order repaired and checked by reading, not compiled, not run on hardware
//
#include <hip/hip_runtime.h>

#define DIM   768
#define SEQ   2048
#define BATCH 8
#define QP    (DIM + 8)   // padded LDS row (bf16) -> 388-dword stride, conflict-free
#define SP    132         // padded S row (f32)
#define PP    136         // padded P row (bf16)

typedef __attribute__((ext_vector_type(16))) __bf16 v16bf;
typedef __attribute__((ext_vector_type(8)))  __bf16 v8bf;
typedef __attribute__((ext_vector_type(8)))  float  v8f;

__device__ __forceinline__ v16bf cat8(v8bf lo, v8bf hi) {
  return __builtin_shufflevector(lo, hi, 0,1,2,3,4,5,6,7,8,9,10,11,12,13,14,15);
}

__device__ __forceinline__ v8bf cvt8(const float4 a, const float4 b) {
  v8bf r;
  r[0]=(__bf16)a.x; r[1]=(__bf16)a.y; r[2]=(__bf16)a.z; r[3]=(__bf16)a.w;
  r[4]=(__bf16)b.x; r[5]=(__bf16)b.y; r[6]=(__bf16)b.z; r[7]=(__bf16)b.w;
  return r;
}

// ---------------------------------------------------------------------------
// Pass 0: f32 -> bf16 bulk conversion (8 elems/thread, 16B stores)
// ---------------------------------------------------------------------------
__global__ __launch_bounds__(256)
void cvt_bf16(const float* __restrict__ src, __bf16* __restrict__ dst, int n8)
{
  const int i = blockIdx.x * 256 + threadIdx.x;
  if (i >= n8) return;
  const float4* s = (const float4*)src + (size_t)i * 2;
  *(v8bf*)(dst + (size_t)i * 8) = cvt8(s[0], s[1]);
}

// ---------------------------------------------------------------------------
// Pass 1: Y = x @ W^T (z selects Wq/Wk/Wv), all-bf16 operands, f32 accum.
// One wave computes a 16x128 tile: per K-step 2 A-loads + 16 B-loads, 8 WMMAs.
// Q pre-scaled by 1/sqrt(768); V stored transposed (b, d, j) via LDS staging.
// ---------------------------------------------------------------------------
__global__ __launch_bounds__(32)
void gemm_qkv(const __bf16* __restrict__ xb,
              const __bf16* __restrict__ Wb,   // [Wq | Wk | Wv] bf16
              __bf16* __restrict__ Qb,
              __bf16* __restrict__ Kb,
              __bf16* __restrict__ Vt)
{
  __shared__ __align__(16) __bf16 Lt[128][24]; // transposed tile, 48B rows (conflict-free)

  const int lane = threadIdx.x;
  const int col  = lane & 15;
  const int h    = lane >> 4;
  const int m0 = blockIdx.x * 16;
  const int n0 = blockIdx.y * 128;
  const int z  = blockIdx.z;
  const __bf16* W = Wb + (size_t)z * (DIM * DIM);

  v8f acc[8];
  #pragma unroll
  for (int t = 0; t < 8; ++t) acc[t] = (v8f){};

  const __bf16* xrow = xb + (size_t)(m0 + col) * DIM;

  for (int k0 = 0; k0 < DIM; k0 += 32) {
    v8bf a0 = *(const v8bf*)(xrow + k0 + 8 * h);
    v8bf a1 = *(const v8bf*)(xrow + k0 + 16 + 8 * h);
    v8bf bl[8], bh[8];
    #pragma unroll
    for (int t = 0; t < 8; ++t) {
      const __bf16* wr = W + (size_t)(n0 + t * 16 + col) * DIM + k0 + 8 * h;
      bl[t] = *(const v8bf*)wr;
      bh[t] = *(const v8bf*)(wr + 16);
    }
    const v16bf A = cat8(a0, a1);
    #pragma unroll
    for (int t = 0; t < 8; ++t)
      acc[t] = __builtin_amdgcn_wmma_f32_16x16x32_bf16(false, A, false,
                                                       cat8(bl[t], bh[t]),
                                                       (short)0, acc[t], false, false);
  }

  const float qscale = 0.03608439182435161f; // 1/sqrt(768)
  if (z == 0) {
    #pragma unroll
    for (int t = 0; t < 8; ++t)
      #pragma unroll
      for (int v = 0; v < 8; ++v)
        Qb[(size_t)(m0 + v + 8 * h) * DIM + n0 + t * 16 + col] =
            (__bf16)(acc[t][v] * qscale);
  } else if (z == 1) {
    #pragma unroll
    for (int t = 0; t < 8; ++t)
      #pragma unroll
      for (int v = 0; v < 8; ++v)
        Kb[(size_t)(m0 + v + 8 * h) * DIM + n0 + t * 16 + col] = (__bf16)acc[t][v];
  } else {
    // transpose 16x128 tile through LDS, then contiguous 16B runs along j
    #pragma unroll
    for (int t = 0; t < 8; ++t)
      #pragma unroll
      for (int v = 0; v < 8; ++v)
        Lt[t * 16 + col][v + 8 * h] = (__bf16)acc[t][v];
    // single-wave workgroup: LDS ops are in-order, no barrier needed
    const int bb = m0 >> 11, jb = m0 & 2047;
    #pragma unroll
    for (int i = 0; i < 8; ++i) {
      const int q = lane + 32 * i;         // 0..255 chunks of 16B
      const int d = q >> 1, half = q & 1;  // d: 0..127, half: j 0..7 / 8..15
      const uint4 val = *(const uint4*)(&Lt[d][half * 8]);
      *(uint4*)(Vt + ((size_t)bb * DIM + n0 + d) * SEQ + jb + half * 8) = val;
    }
  }
}

// ---------------------------------------------------------------------------
// Pass 2: flash attention. Block = 8 waves = 16 queries of one batch.
// ---------------------------------------------------------------------------
__global__ __launch_bounds__(256)
void attn_kernel(const __bf16* __restrict__ Qb,
                 const __bf16* __restrict__ Kb,
                 const __bf16* __restrict__ Vt,
                 float* __restrict__ out)
{
  __shared__ __align__(16) __bf16 Qs[16][QP];
  __shared__ __align__(16) float  Ss[16][SP];
  __shared__ __align__(16) __bf16 Ps[16][PP];
  __shared__ float mrow[16], lrow[16], arow[16], rs[16];

  const int tid  = threadIdx.x;
  const int lane = tid & 31;
  const int wave = tid >> 5;
  const int col  = lane & 15;
  const int h    = lane >> 4;
  const int q0 = blockIdx.x * 16;
  const int b  = blockIdx.y;

  { // async copy Q tile (16 x 768 bf16) global -> LDS, ASYNCcnt-tracked
    const __bf16* src = Qb + ((size_t)b * SEQ + q0) * DIM;
    for (int i = tid; i < 16 * DIM / 8; i += 256) {
      const int r = i / (DIM / 8), c = i % (DIM / 8);
      const unsigned lds = (unsigned)(size_t)(&Qs[r][c * 8]);
      const unsigned long long ga = (unsigned long long)(size_t)(src + (size_t)i * 8);
      asm volatile("global_load_async_to_lds_b128 %0, %1, off"
                   :: "v"(lds), "v"(ga) : "memory");
    }
    asm volatile("s_wait_asynccnt 0x0" ::: "memory");
  }
  if (tid < 16) { mrow[tid] = -1e30f; lrow[tid] = 0.f; }
  __syncthreads();

  v8f oacc[6];
  #pragma unroll
  for (int t = 0; t < 6; ++t) oacc[t] = (v8f){};
  const int d0 = wave * 96;

  for (int j0 = 0; j0 < SEQ; j0 += 128) {
    // ---- phase 1: S tile (this wave: keys j0 + wave*16 .. +15) ----
    v8f s = (v8f){};
    const __bf16* krow = Kb + ((size_t)b * SEQ + j0 + wave * 16 + col) * DIM;
    #pragma unroll 4
    for (int k0 = 0; k0 < DIM; k0 += 32) {
      v8bf alo = *(const v8bf*)(&Qs[col][k0 + 8 * h]);
      v8bf ahi = *(const v8bf*)(&Qs[col][k0 + 16 + 8 * h]);
      v8bf blo = *(const v8bf*)(krow + k0 + 8 * h);
      v8bf bhi = *(const v8bf*)(krow + k0 + 16 + 8 * h);
      s = __builtin_amdgcn_wmma_f32_16x16x32_bf16(false, cat8(alo, ahi), false,
                                                  cat8(blo, bhi), (short)0, s,
                                                  false, false);
    }
    #pragma unroll
    for (int v = 0; v < 8; ++v) Ss[v + 8 * h][wave * 16 + col] = s[v];
    __syncthreads();

    // ---- phase 2: row max + correction factor; zero row-sum scratch ----
    if (tid < 16) {
      const float mo = mrow[tid];
      float mx = mo;
      for (int j = 0; j < 128; ++j) mx = fmaxf(mx, Ss[tid][j]);
      mrow[tid] = mx;
      arow[tid] = __expf(mo - mx);
      rs[tid] = 0.f;
    }
    __syncthreads();

    // ---- phase 3: P = exp(S - m) (bf16) + per-row sum via LDS float atomics ----
    {
      const int r = tid >> 4, jj = (tid & 15) * 8;  // 8 consecutive cols of one row
      const float mr = mrow[r];
      float sum = 0.f;
      #pragma unroll
      for (int u = 0; u < 8; ++u) {
        const float p = __expf(Ss[r][jj + u] - mr);
        Ps[r][jj + u] = (__bf16)p;
        sum += p;
      }
      atomicAdd(&rs[r], sum);
    }
    __syncthreads();

    // ---- phase 4: l update ----
    if (tid < 16) lrow[tid] = arow[tid] * lrow[tid] + rs[tid];

    // ---- phase 5: rescale O, accumulate P @ V (this wave: d0 .. d0+95) ----
    float al[8];
    #pragma unroll
    for (int v = 0; v < 8; ++v) al[v] = arow[v + 8 * h];
    #pragma unroll
    for (int t = 0; t < 6; ++t)
      #pragma unroll
      for (int v = 0; v < 8; ++v) oacc[t][v] *= al[v];

    #pragma unroll
    for (int kc = 0; kc < 4; ++kc) {
      v8bf alo = *(const v8bf*)(&Ps[col][kc * 32 + 8 * h]);
      v8bf ahi = *(const v8bf*)(&Ps[col][kc * 32 + 16 + 8 * h]);
      v16bf A = cat8(alo, ahi);
      v8bf bl[6], bh[6];
      #pragma unroll
      for (int t = 0; t < 6; ++t) {
        const __bf16* vc = Vt + ((size_t)b * DIM + d0 + t * 16 + col) * SEQ
                              + j0 + kc * 32 + 8 * h;
        bl[t] = *(const v8bf*)vc;
        bh[t] = *(const v8bf*)(vc + 16);
      }
      #pragma unroll
      for (int t = 0; t < 6; ++t)
        oacc[t] = __builtin_amdgcn_wmma_f32_16x16x32_bf16(false, A, false,
                                                          cat8(bl[t], bh[t]), (short)0,
                                                          oacc[t], false, false);
    }
    __syncthreads();
  }

  // ---- epilogue: O / l ----
  float li[8];
  #pragma unroll
  for (int v = 0; v < 8; ++v) li[v] = 1.f / lrow[v + 8 * h];
  #pragma unroll
  for (int t = 0; t < 6; ++t)
    #pragma unroll
    for (int v = 0; v < 8; ++v) {
      const int r = v + 8 * h;
      out[((size_t)b * SEQ + q0 + r) * DIM + d0 + t * 16 + col] = oacc[t][v] * li[v];
    }
}

// ---------------------------------------------------------------------------
extern "C" void kernel_launch(void* const* d_in, const int* in_sizes, int n_in,
                              void* d_out, int out_size, void* d_ws, size_t ws_size,
                              hipStream_t stream) {
  // setup_inputs order: x, Wk, Wq, Wv
  const float* x  = (const float*)d_in[0];
  const float* Wk = (const float*)d_in[1];
  const float* Wq = (const float*)d_in[2];
  const float* Wv = (const float*)d_in[3];
  float* out = (float*)d_out;

  const size_t elems  = (size_t)BATCH * SEQ * DIM;  // 12,582,912
  const size_t welems = (size_t)DIM * DIM;          //    589,824

  __bf16* xb = (__bf16*)d_ws;          // bf16 x
  __bf16* Wb = xb + elems;             // [Wq | Wk | Wv] bf16
  __bf16* Qb = Wb + 3 * welems;
  __bf16* Kb = Qb + elems;
  __bf16* Vt = Kb + elems;

  const int nx8 = (int)(elems / 8), nw8 = (int)(welems / 8);
  cvt_bf16<<<(nx8 + 255) / 256, 256, 0, stream>>>(x,  xb,              nx8);
  cvt_bf16<<<(nw8 + 255) / 256, 256, 0, stream>>>(Wq, Wb + 0 * welems, nw8);
  cvt_bf16<<<(nw8 + 255) / 256, 256, 0, stream>>>(Wk, Wb + 1 * welems, nw8);
  cvt_bf16<<<(nw8 + 255) / 256, 256, 0, stream>>>(Wv, Wb + 2 * welems, nw8);

  dim3 g1(BATCH * SEQ / 16, DIM / 128, 3);
  gemm_qkv<<<g1, 32, 0, stream>>>(xb, Wb, Qb, Kb, Vt);

  dim3 g2(SEQ / 16, BATCH);
  attn_kernel<<<g2, 256, 0, stream>>>(Qb, Kb, Vt, out);
}
